// SelfAttention2_83262236000305
// MI455X (gfx1250) — compile-verified, matrix operand order checked by reading, not run on hardware
//
#include <hip/hip_runtime.h>

// ---------------------------------------------------------------------------
// Self-attention (B=4, C=64, N=4096), flash-attention with CDNA5 WMMA.
//   proj_pack    : q/k/v = W*x+b (fp32), scatter-stored as f16 WMMA fragments
//                  (q pre-scaled by log2(e) so softmax runs in exp2 domain)
//   flash_partial: split-K flash attention. Block = 4 query tiles sharing one
//                  key slice; K/V chunks staged to LDS with
//                  global_load_async_to_lds_b128 (double buffered, ASYNCcnt).
//                  S = Q^T K via v_wmma_f32_16x16x32_f16, online softmax with
//                  v_permlane16 reductions + deferred sums, P x V^T via WMMA.
//   combine      : merge split-K partials, apply gamma*out + x.
// ws: 6 MB fragments + NS*4.3 MB partials (NS picked from ws_size).
// ---------------------------------------------------------------------------

typedef __attribute__((ext_vector_type(16))) _Float16 v16h;
typedef __attribute__((ext_vector_type(8)))  float    v8f;

#define C64 64
#define NB  4
#define NN  4096
#define NT  (NN / 16)    // 256 query/key 16-tiles per batch
#define KC  (NN / 32)    // 128 key chunks of 32 per batch
#define FRAG_ELEMS (NB * NN * C64)   // halves per matrix = 1,048,576
#define PART_FLOATS 1056             // 16x64 acc + 16 m + 16 rowsum
#define LOG2E 1.44269504088896340736f

// Fragment mapping for f16 A (16xK, spatial=M) and B (Kx16, spatial=N):
//   lane = spatial + 16*((K/8)&1),  slot = (K&7) + 8*(K/16)   [K in 0..31]

__global__ __launch_bounds__(128) void proj_pack(
    const float* __restrict__ x,
    const float* __restrict__ Wq, const float* __restrict__ bq,
    const float* __restrict__ Wk, const float* __restrict__ bk,
    const float* __restrict__ Wv, const float* __restrict__ bv,
    _Float16* __restrict__ qf, _Float16* __restrict__ kf, _Float16* __restrict__ vf)
{
    const int n = (blockIdx.x & 31) * 128 + threadIdx.x;   // position 0..4095
    const int o = (blockIdx.x >> 5) & 63;                  // output channel
    const int b = blockIdx.x >> 11;                        // batch

    const float* __restrict__ xb = x + (b * C64) * NN + n;
    const float* __restrict__ wq = Wq + o * C64;
    const float* __restrict__ wk = Wk + o * C64;
    const float* __restrict__ wv = Wv + o * C64;

    float aq = bq[o], ak = bk[o], av = bv[o];
#pragma unroll 8
    for (int c = 0; c < C64; ++c) {
        float xv = xb[c * NN];
        aq = fmaf(wq[c], xv, aq);
        ak = fmaf(wk[c], xv, ak);
        av = fmaf(wv[c], xv, av);
    }

    {   // Q/K fragments: spatial = n, K-dim = channel o (two K=32 chunks)
        const int nt = n >> 4, col = n & 15;
        const int f = o >> 5, cc = o & 31;
        const int lane = col + 16 * ((cc >> 3) & 1);
        const int slot = (cc & 7) + 8 * (cc >> 4);
        const int idx = ((b * NT + nt) * 2 + f) * 512 + lane * 16 + slot;
        qf[idx] = (_Float16)(aq * LOG2E);   // exp2-domain softmax
        kf[idx] = (_Float16)ak;
    }
    {   // V fragments (B of P*V^T): spatial = channel o, K-dim = position n
        const int mc = n >> 5, mm = n & 31;
        const int ct = o >> 4, ccol = o & 15;
        const int lane = ccol + 16 * ((mm >> 3) & 1);
        const int slot = (mm & 7) + 8 * (mm >> 4);
        vf[((b * KC + mc) * 4 + ct) * 512 + lane * 16 + slot] = (_Float16)av;
    }
}

__device__ __forceinline__ v8f wmma_f16(v16h a, v16h b, v8f c) {
    return __builtin_amdgcn_wmma_f32_16x16x32_f16(false, a, false, b,
                                                  (short)0, c, false, false);
}

// xor-butterfly within each 16-lane half on the VALU pipe (v_permlane16_b32)
__device__ __forceinline__ float plane16(float v, unsigned s0, unsigned s1) {
    int iv = __float_as_int(v);
    return __int_as_float(__builtin_amdgcn_permlane16(iv, iv, s0, s1, false, false));
}
__device__ __forceinline__ float half_max(float v) {
    v = fmaxf(v, plane16(v, 0x67452301u, 0xEFCDAB89u));   // xor 1
    v = fmaxf(v, plane16(v, 0x54761032u, 0xDCFE98BAu));   // xor 2
    v = fmaxf(v, plane16(v, 0x32107654u, 0xBA98FEDCu));   // xor 4
    v = fmaxf(v, plane16(v, 0xFEDCBA98u, 0x76543210u));   // xor 8
    return v;
}
__device__ __forceinline__ float half_sum(float v) {
    v += plane16(v, 0x67452301u, 0xEFCDAB89u);
    v += plane16(v, 0x54761032u, 0xDCFE98BAu);
    v += plane16(v, 0x32107654u, 0xBA98FEDCu);
    v += plane16(v, 0xFEDCBA98u, 0x76543210u);
    return v;
}

// gfx1250 async DMA: global -> LDS, tracked by ASYNCcnt
__device__ __forceinline__ void async_b128(unsigned lds_off, const void* gaddr) {
    asm volatile("global_load_async_to_lds_b128 %0, %1, off"
                 :: "v"(lds_off), "v"(gaddr) : "memory");
}
__device__ __forceinline__ void wait_async_le4() {
    asm volatile("s_wait_asynccnt 0x4" ::: "memory");
}

// one K/V chunk = 4096B K frags + 4096B V frags; 128 threads x 4 x 16B
__device__ __forceinline__ void load_chunk(_Float16* dst,
                                           const _Float16* kg, const _Float16* vg,
                                           int tid) {
    const unsigned base = (unsigned)(size_t)dst;   // LDS byte offset (low 32 bits)
    const int off = tid * 16;
    async_b128(base + off,        (const char*)kg + off);
    async_b128(base + off + 2048, (const char*)kg + off + 2048);
    async_b128(base + off + 4096, (const char*)vg + off);
    async_b128(base + off + 6144, (const char*)vg + off + 2048);
}

__global__ __launch_bounds__(128) void flash_partial(
    const _Float16* __restrict__ qf, const _Float16* __restrict__ kf,
    const _Float16* __restrict__ vf, float* __restrict__ part,
    int NS, int NSHIFT)
{
    __shared__ __align__(32) _Float16 kv[2][4096];      // double-buffered K|V chunk
    __shared__ __align__(32) float    lds_p[4][16 * 16]; // packed (e0,e1) dwords

    const int tid  = threadIdx.x;
    const int w    = tid >> 5;
    const int lane = tid & 31;
    const int s    = blockIdx.x & (NS - 1);   // key-slice (shared by all 4 waves)
    const int g    = blockIdx.x >> NSHIFT;
    const int b    = g >> 6;                  // 64 tile-groups per batch
    const int nt   = (g & 63) * 4 + w;        // this wave's query tile
    const int hi   = lane >> 4;
    const int nl   = lane & 15;

    const v16h aq0 = *(const v16h*)(qf + ((b * NT + nt) * 2 + 0) * 512 + lane * 16);
    const v16h aq1 = *(const v16h*)(qf + ((b * NT + nt) * 2 + 1) * 512 + lane * 16);

    v8f acc[4] = {};
    float m[8], rp[8];                        // running max / per-lane partial sums
#pragma unroll
    for (int r = 0; r < 8; ++r) { m[r] = -1e30f; rp[r] = 0.0f; }

    const _Float16* __restrict__ kb = kf + b * NT * 1024;
    const _Float16* __restrict__ vb = vf + b * KC * 2048;
    float* __restrict__ Lp = lds_p[w];

    const int kcn = KC >> NSHIFT;
    const int kc0 = s * kcn, kc1 = kc0 + kcn;

    // prologue: two chunks in flight
    load_chunk(kv[0], kb + kc0 * 2048, vb + kc0 * 2048, tid);
    const int kpre1 = (kc0 + 1 < kc1) ? kc0 + 1 : kc0;
    load_chunk(kv[1], kb + kpre1 * 2048, vb + kpre1 * 2048, tid);
    wait_async_le4();          // chunk kc0 resident
    __syncthreads();

    for (int kc = kc0; kc < kc1; ++kc) {
        _Float16* Kl = kv[(kc - kc0) & 1];
        _Float16* Vl = Kl + 2048;

        const v16h bk00 = *(const v16h*)(Kl +    0 + lane * 16);
        const v16h bk01 = *(const v16h*)(Kl +  512 + lane * 16);
        const v16h bk10 = *(const v16h*)(Kl + 1024 + lane * 16);
        const v16h bk11 = *(const v16h*)(Kl + 1536 + lane * 16);

        v8f s0 = {}, s1 = {};
        s0 = wmma_f16(aq0, bk00, s0);
        s0 = wmma_f16(aq1, bk01, s0);
        s1 = wmma_f16(aq0, bk10, s1);
        s1 = wmma_f16(aq1, bk11, s1);

        // tile row-max (per 16-lane half), wave-uniform rescale check
        float tm8[8];
        bool upd = false;
#pragma unroll
        for (int r = 0; r < 8; ++r) {
            float tm = half_max(fmaxf(s0[r], s1[r]));
            tm8[r] = tm;
            upd |= (tm > m[r]);
        }
        if (__any((int)upd)) {                 // uniform branch: EXEC stays full
#pragma unroll
            for (int r = 0; r < 8; ++r) {
                const float mn = fmaxf(m[r], tm8[r]);
                const float sc = __builtin_amdgcn_exp2f(m[r] - mn);
                m[r] = mn;
                rp[r] *= sc;
                acc[0][r] *= sc; acc[1][r] *= sc;
                acc[2][r] *= sc; acc[3][r] *= sc;
            }
        }

        // exp2 + pack (e0,e1) into one dword per element-pair; stage in LDS
#pragma unroll
        for (int r = 0; r < 8; ++r) {
            const float e0 = __builtin_amdgcn_exp2f(s0[r] - m[r]);
            const float e1 = __builtin_amdgcn_exp2f(s1[r] - m[r]);
            rp[r] += e0 + e1;                  // deferred cross-lane sum
            Lp[(r + 8 * hi) * 16 + nl] =
                __builtin_bit_cast(float, __builtin_amdgcn_cvt_pkrtz(e0, e1));
        }
        asm volatile("" ::: "memory");
        // A-layout P fragment: 32 contiguous bytes of row nl, then de-interleave
        const v16h raw = __builtin_bit_cast(v16h,
            *(const v8f*)(Lp + nl * 16 + 8 * hi));
        const v16h p = __builtin_shufflevector(raw, raw,
            0, 2, 4, 6, 8, 10, 12, 14, 1, 3, 5, 7, 9, 11, 13, 15);
        asm volatile("" ::: "memory");

#pragma unroll
        for (int ct = 0; ct < 4; ++ct) {
            const v16h bvv = *(const v16h*)(Vl + ct * 512 + lane * 16);
            acc[ct] = wmma_f16(p, bvv, acc[ct]);
        }

        // rotate double buffer: all waves done reading -> refill with chunk kc+2
        __syncthreads();
        int kn = kc + 2; if (kn >= kc1) kn = kc1 - 1;
        load_chunk(Kl, kb + kn * 2048, vb + kn * 2048, tid);
        wait_async_le4();      // chunk kc+1 resident
        __syncthreads();
    }

    // write partial: 16x64 acc (unnormalized), 16 m, 16 rowsum
    float* __restrict__ pb = part + (size_t)((b * NT + nt) * NS + s) * PART_FLOATS;
#pragma unroll
    for (int ct = 0; ct < 4; ++ct)
#pragma unroll
        for (int r = 0; r < 8; ++r)
            pb[(r + 8 * hi) * 64 + ct * 16 + nl] = acc[ct][r];

    float rs[8];
#pragma unroll
    for (int r = 0; r < 8; ++r) rs[r] = half_sum(rp[r]);
    if (nl == 0) {
#pragma unroll
        for (int r = 0; r < 8; ++r) {
            pb[1024 + r + 8 * hi] = m[r];
            pb[1040 + r + 8 * hi] = rs[r];
        }
    }
}

__global__ __launch_bounds__(128) void combine(
    const float* __restrict__ part, const float* __restrict__ x,
    const float* __restrict__ gamma, float* __restrict__ out, int NS)
{
    const int t  = blockIdx.x;            // 0..1023 = (batch, query tile)
    const int b  = t >> 8;
    const int nt = t & 255;
    const int n  = threadIdx.x & 15;      // query row in tile
    const int cg = threadIdx.x >> 4;      // channel group 0..7

    const float* __restrict__ pb = part + (size_t)t * NS * PART_FLOATS;
    const float g = gamma[0];

    float ms[4], rss[4];
#pragma unroll
    for (int s = 0; s < 4; ++s) {
        ms[s]  = (s < NS) ? pb[s * PART_FLOATS + 1024 + n] : -1e30f;
        rss[s] = (s < NS) ? pb[s * PART_FLOATS + 1040 + n] : 0.0f;
    }
    const float M = fmaxf(fmaxf(ms[0], ms[1]), fmaxf(ms[2], ms[3]));
    float wgt[4], denom = 0.0f;
#pragma unroll
    for (int s = 0; s < 4; ++s) {
        wgt[s] = (s < NS) ? __builtin_amdgcn_exp2f(ms[s] - M) : 0.0f;
        denom += rss[s] * wgt[s];
    }
    const float inv = 1.0f / denom;

#pragma unroll
    for (int i = 0; i < 8; ++i) {
        const int c = cg + i * 8;
        float num = 0.0f;
#pragma unroll
        for (int s = 0; s < 4; ++s)
            if (s < NS) num += pb[s * PART_FLOATS + n * 64 + c] * wgt[s];
        const int idx = (b * C64 + c) * NN + nt * 16 + n;
        out[idx] = g * num * inv + x[idx];
    }
}

extern "C" void kernel_launch(void* const* d_in, const int* in_sizes, int n_in,
                              void* d_out, int out_size, void* d_ws, size_t ws_size,
                              hipStream_t stream) {
    (void)in_sizes; (void)n_in; (void)out_size;
    const float* x     = (const float*)d_in[0];
    const float* Wq    = (const float*)d_in[1];
    const float* bq    = (const float*)d_in[2];
    const float* Wk    = (const float*)d_in[3];
    const float* bk    = (const float*)d_in[4];
    const float* Wv    = (const float*)d_in[5];
    const float* bv    = (const float*)d_in[6];
    const float* gamma = (const float*)d_in[7];
    float* out = (float*)d_out;

    _Float16* qf = (_Float16*)d_ws;          // 2 MB
    _Float16* kf = qf + FRAG_ELEMS;          // 2 MB
    _Float16* vf = kf + FRAG_ELEMS;          // 2 MB
    float* part  = (float*)(vf + FRAG_ELEMS);

    const size_t frag_bytes = (size_t)3 * FRAG_ELEMS * sizeof(_Float16);
    int NS = 4, NSHIFT = 2;                   // split-K factor for occupancy
    while (NS > 1 &&
           frag_bytes + (size_t)NB * NT * NS * PART_FLOATS * 4 > ws_size) {
        NS >>= 1; NSHIFT--;
    }

    proj_pack<<<NB * C64 * (NN / 128), 128, 0, stream>>>(x, Wq, bq, Wk, bk, Wv, bv,
                                                         qf, kf, vf);
    flash_partial<<<NB * (NT / 4) * NS, 128, 0, stream>>>(qf, kf, vf, part,
                                                          NS, NSHIFT);
    combine<<<NB * NT, 128, 0, stream>>>(part, x, gamma, out, NS);
}
